// LSTMCell_6150393167883
// MI455X (gfx1250) — matrix operand order repaired and checked by reading, not compiled, not run on hardware
//
#include <hip/hip_runtime.h>

// ---------------------------------------------------------------------------
// LSTM cell on gfx1250 (MI455X): single fused GEMM  z = [u|h] @ [W|U]^T + b
// via v_wmma_f32_16x16x32_f16. Weights tile (shared by all 8 waves of the
// block) staged in LDS, double-buffered, global->LDS prefetch distance 2;
// epilogue fused in-register.
// ---------------------------------------------------------------------------

typedef _Float16 v16h __attribute__((ext_vector_type(16)));
typedef _Float16 v8h  __attribute__((ext_vector_type(8)));
typedef float    v8f  __attribute__((ext_vector_type(8)));

#define BROWS 16384
#define IDIM  512
#define HDIM  512
#define KTOT  1024          // I + H
#define NGATE 4

// -------- conversion: A_f16[16384][1024] = [u | h] in f16 ------------------
__global__ __launch_bounds__(256)
void cvt_inputs(const float* __restrict__ u, const float* __restrict__ h,
                _Float16* __restrict__ A) {
    int tid = blockIdx.x * blockDim.x + threadIdx.x;   // 16384*128 threads
    int row = tid >> 7;
    int c8  = (tid & 127) << 3;                        // k offset, mult of 8
    const float* src = (c8 < IDIM)
        ? (u + (size_t)row * IDIM + c8)
        : (h + (size_t)row * HDIM + (c8 - IDIM));
    float4 a = ((const float4*)src)[0];
    float4 b = ((const float4*)src)[1];
    v8h o = { (_Float16)a.x, (_Float16)a.y, (_Float16)a.z, (_Float16)a.w,
              (_Float16)b.x, (_Float16)b.y, (_Float16)b.z, (_Float16)b.w };
    *(v8h*)(A + (size_t)row * KTOT + c8) = o;
}

// -------- conversion: Wc_f16[2048][1024] = gate-concat of [W | U] ----------
__global__ __launch_bounds__(256)
void cvt_weights(const float* __restrict__ Wi, const float* __restrict__ Wf,
                 const float* __restrict__ Wg, const float* __restrict__ Wo,
                 const float* __restrict__ Ui, const float* __restrict__ Uf,
                 const float* __restrict__ Ug, const float* __restrict__ Uo,
                 _Float16* __restrict__ Wc) {
    int tid  = blockIdx.x * blockDim.x + threadIdx.x;  // 2048*128 threads
    int n    = tid >> 7;
    int c8   = (tid & 127) << 3;
    int gate = n >> 9;
    int rg   = n & 511;
    const float* src;
    if (c8 < IDIM) {
        const float* base = (gate == 0) ? Wi : (gate == 1) ? Wf
                          : (gate == 2) ? Wg : Wo;
        src = base + (size_t)rg * IDIM + c8;
    } else {
        const float* base = (gate == 0) ? Ui : (gate == 1) ? Uf
                          : (gate == 2) ? Ug : Uo;
        src = base + (size_t)rg * HDIM + (c8 - IDIM);
    }
    float4 a = ((const float4*)src)[0];
    float4 b = ((const float4*)src)[1];
    v8h o = { (_Float16)a.x, (_Float16)a.y, (_Float16)a.z, (_Float16)a.w,
              (_Float16)b.x, (_Float16)b.y, (_Float16)b.z, (_Float16)b.w };
    *(v8h*)(Wc + (size_t)n * KTOT + c8) = o;
}

__device__ __forceinline__ float fsigmoid(float x) {
    return 1.0f / (1.0f + __expf(-x));
}

__device__ __forceinline__ v16h mkfrag(v8h lo, v8h hi) {
    return __builtin_shufflevector(lo, hi,
        0, 1, 2, 3, 4, 5, 6, 7, 8, 9, 10, 11, 12, 13, 14, 15);
}

// -------- fused GEMM + LSTM epilogue ---------------------------------------
// block = 256 threads = 8 waves, all with the SAME j0 -> B tile shared 8x.
// Wave owns 64 rows x 16 hidden-cols x 4 gates (16 v8f accumulators).
// grid  = (16384/512, 512/16) = (32, 32)
__global__ __launch_bounds__(256)
void lstm_gemm(const _Float16* __restrict__ A,  const _Float16* __restrict__ Wc,
               const float* __restrict__ c_in,
               const float* __restrict__ bWi, const float* __restrict__ bWf,
               const float* __restrict__ bWg, const float* __restrict__ bWo,
               const float* __restrict__ bi,  const float* __restrict__ bf_,
               const float* __restrict__ bg,  const float* __restrict__ bo,
               float* __restrict__ h_out, float* __restrict__ c_out) {
    // B tile for one K-step: 64 rows (4 gates x 16 cols) x 32 k, double-buffered
    __shared__ __attribute__((aligned(16))) _Float16 ldsB[2][64][32];

    const int tid  = threadIdx.x;
    const int lane = tid & 31;
    const int wave = tid >> 5;
    const int lrow = lane & 15;
    const int half = lane >> 4;

    const int mBase = blockIdx.x * 512 + wave * 64;   // 4 M-tiles of 16
    const int j0    = blockIdx.y * 16;                // hidden column slice

    // --- staging map: one b128 per thread fills the whole 4 KB B tile ---
    const int sRow  = tid >> 2;          // 0..63 = gate*16 + col
    const int sSeg  = (tid & 3) * 8;     // k sub-offset 0,8,16,24
    const int sGate = sRow >> 4;
    const int sCol  = sRow & 15;
    const _Float16* gBsrc =
        Wc + (size_t)(sGate * 512 + j0 + sCol) * KTOT + sSeg;

    v8f acc[NGATE][4];
    #pragma unroll
    for (int g = 0; g < NGATE; ++g)
        #pragma unroll
        for (int t = 0; t < 4; ++t)
            acc[g][t] = (v8f)(0.0f);

    // A 16x32 f16 fragment: lanes 0-15 K={0..7,16..23}; 16-31 K={8..15,24..31}
    const _Float16* aBase = A + (size_t)(mBase + lrow) * KTOT + half * 8;

    // prologue: tile 0 -> LDS buf0; prefetch tile 1 (k=32) into stReg.
    *(v8h*)(&ldsB[0][sRow][sSeg]) = *(const v8h*)(gBsrc);
    v8h stReg = *(const v8h*)(gBsrc + 32);
    __syncthreads();

    // Invariant at entry: stReg holds B tile for (kk + 32).
    auto do_step = [&](int kk, const _Float16 (*bufR)[32],
                       _Float16 (*bufW)[32], bool doStore,
                       bool doPref, int kPref) {
        v8h nxt;
        if (doPref) nxt = *(const v8h*)(gBsrc + kPref);  // tile kk+64, early

        // A fragments: direct global loads (unique rows per wave)
        v16h afrag[4];
        #pragma unroll
        for (int t = 0; t < 4; ++t) {
            const _Float16* p = aBase + (size_t)(t * 16) * KTOT + kk;
            afrag[t] = mkfrag(*(const v8h*)(p), *(const v8h*)(p + 16));
        }

        // B fragments: all 4 gates up front (t-outer WMMA order below keeps
        // them simultaneously live -> distinct octets, one dscnt wait).
        v16h bfr[NGATE];
        #pragma unroll
        for (int g = 0; g < NGATE; ++g) {
            const _Float16* bp = &bufR[g * 16 + lrow][half * 8];
            bfr[g] = mkfrag(*(const v8h*)(bp), *(const v8h*)(bp + 16));
        }

        #pragma unroll
        for (int t = 0; t < 4; ++t) {
            #pragma unroll
            for (int g = 0; g < NGATE; ++g) {
                acc[g][t] = __builtin_amdgcn_wmma_f32_16x16x32_f16(
                    false, afrag[t], false, bfr[g],
                    (short)0, acc[g][t], false, false);
            }
        }

        // store tile kk+32 (loaded a full K-step ago -> wait is free)
        if (doStore) *(v8h*)(&bufW[sRow][sSeg]) = stReg;
        __syncthreads();   // one barrier per K-step (double buffer => no WAR)
        if (doPref) stReg = nxt;
    };

    for (int k = 0; k < KTOT; k += 64) {
        do_step(k,      ldsB[0], ldsB[1], true,
                (k + 64) < KTOT, k + 64);
        do_step(k + 32, ldsB[1], ldsB[0], (k + 64) < KTOT,
                (k + 96) < KTOT, k + 96);
    }

    // ---- epilogue: bias + activations + recurrence, all in-register ----
    const int jg = j0 + lrow;                 // this lane's hidden index
    const float Bi = bWi[jg] + bi[jg];
    const float Bf = bWf[jg] + bf_[jg];
    const float Bg = bWg[jg] + bg[jg];
    const float Bo = bWo[jg] + bo[jg];

    #pragma unroll
    for (int t = 0; t < 4; ++t) {
        #pragma unroll
        for (int r = 0; r < 8; ++r) {
            // C/D layout: vgpr r -> M = r (lanes 0-15) or r+8 (lanes 16-31)
            const int m = mBase + t * 16 + half * 8 + r;
            const size_t idx = (size_t)m * HDIM + jg;
            float ig = fsigmoid(acc[0][t][r] + Bi);
            float fg = fsigmoid(acc[1][t][r] + Bf);
            float gg = tanhf(acc[2][t][r] + Bg);
            float og = fsigmoid(acc[3][t][r] + Bo);
            float cn = fg * c_in[idx] + ig * gg;
            h_out[idx] = og * tanhf(cn);
            c_out[idx] = cn;
        }
    }
}

extern "C" void kernel_launch(void* const* d_in, const int* in_sizes, int n_in,
                              void* d_out, int out_size, void* d_ws, size_t ws_size,
                              hipStream_t stream) {
    const float* u   = (const float*)d_in[0];
    const float* h   = (const float*)d_in[1];
    const float* c   = (const float*)d_in[2];
    const float* Wi  = (const float*)d_in[3];
    const float* bWi = (const float*)d_in[4];
    const float* Wf  = (const float*)d_in[5];
    const float* bWf = (const float*)d_in[6];
    const float* Wg  = (const float*)d_in[7];
    const float* bWg = (const float*)d_in[8];
    const float* Wo  = (const float*)d_in[9];
    const float* bWo = (const float*)d_in[10];
    const float* Ui  = (const float*)d_in[11];
    const float* Uf  = (const float*)d_in[12];
    const float* Ug  = (const float*)d_in[13];
    const float* Uo  = (const float*)d_in[14];
    const float* bi  = (const float*)d_in[15];
    const float* bf_ = (const float*)d_in[16];
    const float* bg  = (const float*)d_in[17];
    const float* bo  = (const float*)d_in[18];

    float* h_out = (float*)d_out;                       // h_next first
    float* c_out = h_out + (size_t)BROWS * HDIM;        // then c_next

    _Float16* Af16 = (_Float16*)d_ws;                        // 32 MB
    _Float16* Wf16 = Af16 + (size_t)BROWS * KTOT;            // + 4 MB

    cvt_inputs<<<(BROWS * (KTOT / 8)) / 256, 256, 0, stream>>>(u, h, Af16);
    cvt_weights<<<(NGATE * HDIM * (KTOT / 8)) / 256, 256, 0, stream>>>(
        Wi, Wf, Wg, Wo, Ui, Uf, Ug, Uo, Wf16);

    dim3 grid(BROWS / 512, HDIM / 16);   // (32, 32)
    lstm_gemm<<<grid, 256, 0, stream>>>(Af16, Wf16, c,
                                        bWi, bWf, bWg, bWo,
                                        bi, bf_, bg, bo,
                                        h_out, c_out);
}
